// TreeFFN_45981919871645
// MI455X (gfx1250) — compile-verified, matrix operand order checked by reading, not run on hardware
//
#include <hip/hip_runtime.h>
#include <cstdint>
#include <cstddef>

// ---------------------------------------------------------------------------
// TreeFFN on MI455X (gfx1250, wave32).
//   h   = X @ Ws.T                         (bf16 WMMA, f32 accum)
//   10x: agg = scatter_add(gather(h_bf16)) (f32 hardware float atomics)
//        h   = relu(agg @ Wpc.T + h) + h   (bf16 WMMA, A cast in-register)
//        out += sigmoid(T - it) * h
// Grid order: column-groups in blockIdx.x (8, co-resident) so the 205MB
// A matrix streams through L2 once per GEMM instead of 8x. Each wave owns
// two 16-row strips so every B fragment from LDS feeds two WMMAs.
// ---------------------------------------------------------------------------

#define D_IN 256
#define D_H  512
#define MAX_ITERS 10

typedef __attribute__((ext_vector_type(16))) __bf16 v16bf;
typedef __attribute__((ext_vector_type(8)))  float  v8f;

union FragBF {
  v16bf v;
  uint4 u[2];
};

__device__ __forceinline__ unsigned short f2bf(float f) {
  // round-to-nearest-even f32 -> bf16 (bit exact, used for stores)
  unsigned int u = __float_as_uint(f);
  unsigned int lsb = (u >> 16) & 1u;
  u += 0x7fffu + lsb;
  return (unsigned short)(u >> 16);
}

__device__ __forceinline__ float bflo(unsigned int u) {
  return __uint_as_float(u << 16);
}
__device__ __forceinline__ float bfhi(unsigned int u) {
  return __uint_as_float(u & 0xffff0000u);
}

// Load one 16x32 A fragment from f32 memory, converting to bf16 in-register
// (lowers to v_cvt_pk_bf16_f32). Layout: lane&15 = M row, lanes 16-31 carry
// the K+8 halves (elements 0..7 = K+0..7, 8..15 = K+16..23 relative to ko).
__device__ __forceinline__ FragBF load_a_f32(const float* __restrict__ p) {
  float4 f0 = *(const float4*)(p);
  float4 f1 = *(const float4*)(p + 4);
  float4 f2 = *(const float4*)(p + 16);
  float4 f3 = *(const float4*)(p + 20);
  FragBF a;
  a.v[0]  = (__bf16)f0.x; a.v[1]  = (__bf16)f0.y;
  a.v[2]  = (__bf16)f0.z; a.v[3]  = (__bf16)f0.w;
  a.v[4]  = (__bf16)f1.x; a.v[5]  = (__bf16)f1.y;
  a.v[6]  = (__bf16)f1.z; a.v[7]  = (__bf16)f1.w;
  a.v[8]  = (__bf16)f2.x; a.v[9]  = (__bf16)f2.y;
  a.v[10] = (__bf16)f2.z; a.v[11] = (__bf16)f2.w;
  a.v[12] = (__bf16)f3.x; a.v[13] = (__bf16)f3.y;
  a.v[14] = (__bf16)f3.z; a.v[15] = (__bf16)f3.w;
  return a;
}

// Cast n floats -> bf16 (n % 8 == 0). Used once for the small weights.
__global__ void cast_f32_bf16(const float* __restrict__ src,
                              unsigned short* __restrict__ dst,
                              long long n) {
  long long i = ((long long)blockIdx.x * blockDim.x + threadIdx.x) * 8;
  if (i >= n) return;
  float4 a = *(const float4*)(src + i);
  float4 b = *(const float4*)(src + i + 4);
  uint4 o;
  o.x = (unsigned)f2bf(a.x) | ((unsigned)f2bf(a.y) << 16);
  o.y = (unsigned)f2bf(a.z) | ((unsigned)f2bf(a.w) << 16);
  o.z = (unsigned)f2bf(b.x) | ((unsigned)f2bf(b.y) << 16);
  o.w = (unsigned)f2bf(b.z) | ((unsigned)f2bf(b.w) << 16);
  *(uint4*)(dst + i) = o;
}

// msg = h[p] + h[c] (bf16 gather, widened to f32); agg[p]+=msg; agg[c]+=msg.
// 64 threads per edge, 8 columns each (one b128 bf16 load per endpoint).
__global__ void edge_scatter(const unsigned short* __restrict__ hb,
                             const int* __restrict__ pidx,
                             const int* __restrict__ cidx,
                             float* __restrict__ agg,
                             int E) {
  long long gid = (long long)blockIdx.x * blockDim.x + threadIdx.x;
  int c8 = (int)(gid & 63);           // D_H/8 = 64 chunks per row
  long long e = gid >> 6;
  if (e >= E) return;
  long long p = pidx[e];
  long long c = cidx[e];
  uint4 up = *(const uint4*)(hb + p * D_H + c8 * 8);
  uint4 uc = *(const uint4*)(hb + c * D_H + c8 * 8);
  float m0 = bflo(up.x) + bflo(uc.x), m1 = bfhi(up.x) + bfhi(uc.x);
  float m2 = bflo(up.y) + bflo(uc.y), m3 = bfhi(up.y) + bfhi(uc.y);
  float m4 = bflo(up.z) + bflo(uc.z), m5 = bfhi(up.z) + bfhi(uc.z);
  float m6 = bflo(up.w) + bflo(uc.w), m7 = bfhi(up.w) + bfhi(uc.w);
  float* ap = agg + p * D_H + c8 * 8;
  float* ac = agg + c * D_H + c8 * 8;
#define ATOM(P, V) \
  __hip_atomic_fetch_add((P), (V), __ATOMIC_RELAXED, __HIP_MEMORY_SCOPE_AGENT)
  ATOM(ap + 0, m0); ATOM(ap + 1, m1); ATOM(ap + 2, m2); ATOM(ap + 3, m3);
  ATOM(ap + 4, m4); ATOM(ap + 5, m5); ATOM(ap + 6, m6); ATOM(ap + 7, m7);
  ATOM(ac + 0, m0); ATOM(ac + 1, m1); ATOM(ac + 2, m2); ATOM(ac + 3, m3);
  ATOM(ac + 4, m4); ATOM(ac + 5, m5); ATOM(ac + 6, m6); ATOM(ac + 7, m7);
#undef ATOM
}

// D[M,512] = A[M,K](f32, cast to bf16 in-register) @ W[512,K](bf16)^T.
// Block: 160 threads = 5 waves. Wave w owns rows [ (by*5+w)*32 , +32 )
// (two 16-row strips) and cols [ bx*64 , +64 ) as 2x4 WMMA 16x16 tiles.
// W staged via LDS in 256-element K slices (row pad 8 bf16 ->
// conflict-free ds_load_b128); each B fragment feeds two WMMAs.
// Grid covers M exactly => EXEC all-1s for every WMMA (ISA requirement).
template <int K, bool UPDATE>
__global__ __launch_bounds__(160) void wmma_gemm(
    const float* __restrict__ A,
    const unsigned short* __restrict__ W,
    float* __restrict__ H,           // h (f32): read residual when UPDATE
    unsigned short* __restrict__ HB, // bf16 mirror of h for the gather
    float* __restrict__ OUT,         // acc accumulator (UPDATE only)
    const float* __restrict__ Tptr,  // temperature scalar (UPDATE only)
    int stepIdx) {
  constexpr int STAGE = 256;
  constexpr int LROW = STAGE + 8;    // 264 ushorts: bank-conflict pad
  __shared__ unsigned short ldsW[64 * LROW];

  const int tid  = threadIdx.x;
  const int wave = tid >> 5;
  const int lane = tid & 31;
  const int n0 = blockIdx.x * 64;                 // column group (8 groups)
  const int m0 = (blockIdx.y * 5 + wave) * 32;    // 32-row double strip
  const int ko = (lane >> 4) * 8;                 // lanes 16-31 carry K+8

  v8f acc[2][4] = {};

  for (int kb0 = 0; kb0 < K; kb0 += STAGE) {
    if (kb0) __syncthreads();        // protect LDS re-fill
    for (int c = tid; c < 64 * (STAGE / 8); c += 160) {
      int r  = c >> 5;               // 32 uint4 chunks per row
      int cc = (c & 31) * 8;
      *(uint4*)&ldsW[r * LROW + cc] =
          *(const uint4*)&W[(size_t)(n0 + r) * K + kb0 + cc];
    }
    __syncthreads();

    const float* ap0 = A + (size_t)(m0 + (lane & 15)) * K + kb0 + ko;
    const float* ap1 = ap0 + (size_t)16 * K;
    if (kb0 + STAGE < K) {
      __builtin_prefetch(ap0 + STAGE, 0, 1);
      __builtin_prefetch(ap1 + STAGE, 0, 1);
    }
#pragma unroll
    for (int kk = 0; kk < STAGE; kk += 32) {
      FragBF a0 = load_a_f32(ap0 + kk);
      FragBF a1 = load_a_f32(ap1 + kk);
#pragma unroll
      for (int j = 0; j < 4; ++j) {
        FragBF b;  // lane&15 = output column within tile = W row
        const unsigned short* bp =
            &ldsW[(size_t)(j * 16 + (lane & 15)) * LROW + kk + ko];
        b.u[0] = *(const uint4*)bp;
        b.u[1] = *(const uint4*)(bp + 16);
        acc[0][j] = __builtin_amdgcn_wmma_f32_16x16x32_bf16(
            false, a0.v, false, b.v, (short)0, acc[0][j], false, false);
        acc[1][j] = __builtin_amdgcn_wmma_f32_16x16x32_bf16(
            false, a1.v, false, b.v, (short)0, acc[1][j], false, false);
      }
    }
  }

  // Epilogue. C/D layout: VGPR v -> row base + v (+8 for lanes 16-31),
  // lane&15 -> column within a 16-wide tile.
  float sw = 0.0f;
  if (UPDATE) {
    float T = *Tptr;
    sw = 1.0f / (1.0f + __expf((float)stepIdx - T));  // sigmoid(T - step)
  }
  const int cbase = n0 + (lane & 15);
#pragma unroll
  for (int s = 0; s < 2; ++s) {
    const int rbase = m0 + s * 16 + ((lane >> 4) << 3);
#pragma unroll
    for (int j = 0; j < 4; ++j) {
#pragma unroll
      for (int v = 0; v < 8; ++v) {
        size_t idx = (size_t)(rbase + v) * D_H + (cbase + j * 16);
        if (UPDATE) {
          float hv = H[idx];
          float t  = acc[s][j][v] + hv;    // agg @ Wpc.T + h
          float r  = fmaxf(t, 0.0f) + hv;  // relu(.) + h  (residual)
          H[idx]   = r;                    // h <- step_h (f32)
          HB[idx]  = f2bf(r);              // bf16 mirror for next gather
          OUT[idx] += sw * r;              // acc += step_w * step_h
        } else {
          H[idx]  = acc[s][j][v];
          HB[idx] = f2bf(acc[s][j][v]);
        }
      }
    }
  }
}

extern "C" void kernel_launch(void* const* d_in, const int* in_sizes, int n_in,
                              void* d_out, int out_size, void* d_ws,
                              size_t ws_size, hipStream_t stream) {
  (void)n_in; (void)ws_size;
  const float* node = (const float*)d_in[0];   // [N, 256] f32
  const int*   ei   = (const int*)d_in[1];     // [2, E]
  const float* Tptr = (const float*)d_in[2];   // scalar
  const float* Ws   = (const float*)d_in[3];   // [512, 256] f32
  const float* Wpc  = (const float*)d_in[4];   // [512, 512] f32
  float* out = (float*)d_out;                  // [N, 512] f32

  const long long N = in_sizes[0] / D_IN;      // 100000
  const long long E = in_sizes[1] / 2;         // 99999

  auto al = [](size_t x) { return (x + 255) & ~(size_t)255; };
  char* w = (char*)d_ws;
  float* h   = (float*)w;                     w += al((size_t)N * D_H * 4);
  float* agg = (float*)w;                     w += al((size_t)N * D_H * 4);
  unsigned short* hb   = (unsigned short*)w;  w += al((size_t)N * D_H * 2);
  unsigned short* wsb  = (unsigned short*)w;  w += al((size_t)D_H * D_IN * 2);
  unsigned short* wpcb = (unsigned short*)w;  w += al((size_t)D_H * D_H * 2);

  hipMemsetAsync(d_out, 0, (size_t)out_size * sizeof(float), stream);

  {
    long long n = (long long)D_H * D_IN;
    cast_f32_bf16<<<(unsigned)((n / 8 + 255) / 256), 256, 0, stream>>>(Ws, wsb, n);
    n = (long long)D_H * D_H;
    cast_f32_bf16<<<(unsigned)((n / 8 + 255) / 256), 256, 0, stream>>>(Wpc, wpcb, n);
  }

  // x = column group (8, co-resident for L2 reuse of A), y = row groups.
  // Each block covers 5 waves * 32 rows = 160 rows; N % 160 == 0.
  dim3 ggrid(D_H / 64, (unsigned)(N / 160));
  wmma_gemm<D_IN, false><<<ggrid, 160, 0, stream>>>(node, wsb, h, hb, nullptr,
                                                    nullptr, 0);

  const unsigned eblocks = (unsigned)((E * 64 + 255) / 256);
  for (int it = 0; it < MAX_ITERS; ++it) {
    hipMemsetAsync(agg, 0, (size_t)N * D_H * 4, stream);
    edge_scatter<<<eblocks, 256, 0, stream>>>(hb, ei, ei + E, agg, (int)E);
    wmma_gemm<D_H, true><<<ggrid, 160, 0, stream>>>(agg, wpcb, h, hb, out,
                                                    Tptr, it);
  }
}
